// CRF_29815662969206
// MI455X (gfx1250) — matrix operand order repaired and checked by reading, not compiled, not run on hardware
//
#include <hip/hip_runtime.h>

// CRF loss for MI455X (gfx1250, wave32).
// Forward algorithm recast as: alpha_new = obs + c + log( exp(alpha - c) @ exp(trans) )
// so the 34x34 log-semiring contraction runs on v_wmma_f32_16x16x32_f16.
// One wave handles 16 batch rows; states padded 34 -> K=64 / N=48.
// Emission tiles are double-buffered in LDS via async global->LDS copies.
// Cross-lane reductions use DPP16; all log/exp are raw v_log_f32/v_exp_f32
// (base-2 domain internally, converted to natural log once at the end).
// Pad row K=63 carries a 2^-14 bias column so log2(D) is always finite.

typedef _Float16 v16h __attribute__((ext_vector_type(16)));
typedef float    v8f  __attribute__((ext_vector_type(8)));
typedef int      i32x4 __attribute__((vector_size(16)));   // matches builtin param type

typedef __attribute__((address_space(1))) i32x4 gl_i32x4;  // global (device) b128
typedef __attribute__((address_space(3))) i32x4 ls_i32x4;  // LDS (shared) b128

#define B_DIM 512
#define L_DIM 512
#define T_DIM 32
#define S_DIM 34              // 0=START, 1..32=tags, 33=STOP
#define NEG_V (-1000.0f)
#define PAD_V (-1.0e30f)
#define LOG2E 1.44269504088896340736f
#define LN2   0.69314718055994530942f
#define FLOORH 6.103515625e-5f          // 2^-14, exact in f16 (min normal)

#ifndef __has_builtin
#define __has_builtin(x) 0
#endif

#if __has_builtin(__builtin_amdgcn_global_load_async_to_lds_b128)
#define HAVE_ASYNC 1
#else
#define HAVE_ASYNC 0
#endif

#if HAVE_ASYNC
#if __has_builtin(__builtin_amdgcn_s_wait_asynccnt)
#define WAIT_ASYNC_4() __builtin_amdgcn_s_wait_asynccnt(4)
#define WAIT_ASYNC_0() __builtin_amdgcn_s_wait_asynccnt(0)
#else
#define WAIT_ASYNC_4() asm volatile("s_wait_asynccnt 4" ::: "memory")
#define WAIT_ASYNC_0() asm volatile("s_wait_asynccnt 0" ::: "memory")
#endif
#else
#define WAIT_ASYNC_4()
#define WAIT_ASYNC_0()
#endif

// Raw hardware exp2/log2 (v_exp_f32 / v_log_f32): inputs here are guaranteed
// normal-range, so skip libm's denormal guards entirely.
#if __has_builtin(__builtin_amdgcn_exp2f)
#define EXP2F(x) __builtin_amdgcn_exp2f(x)
#else
#define EXP2F(x) exp2f(x)
#endif
#if __has_builtin(__builtin_amdgcn_logf)
#define LOG2F(x) __builtin_amdgcn_logf(x)
#else
#define LOG2F(x) __log2f(x)
#endif

// ---- 16-lane cross-lane reductions via DPP16 (VALU, no LDS round-trips) ----
#if __has_builtin(__builtin_amdgcn_update_dpp)
#define DPP_F(v, ctrl) \
  __int_as_float(__builtin_amdgcn_update_dpp(0, __float_as_int(v), (ctrl), 0xf, 0xf, true))
__device__ __forceinline__ float redmax16(float v) {
  v = fmaxf(v, DPP_F(v, 0xB1));   // quad_perm [1,0,3,2] : xor 1
  v = fmaxf(v, DPP_F(v, 0x4E));   // quad_perm [2,3,0,1] : xor 2
  v = fmaxf(v, DPP_F(v, 0x141));  // row_half_mirror     : xor 7 (within 8)
  v = fmaxf(v, DPP_F(v, 0x140));  // row_mirror          : xor 15 (within 16)
  return v;
}
__device__ __forceinline__ float redsum16(float v) {
  v += DPP_F(v, 0xB1);
  v += DPP_F(v, 0x4E);
  v += DPP_F(v, 0x141);
  v += DPP_F(v, 0x140);
  return v;
}
#else
__device__ __forceinline__ float redmax16(float v) {
  for (int off = 1; off < 16; off <<= 1) v = fmaxf(v, __shfl_xor(v, off, 16));
  return v;
}
__device__ __forceinline__ float redsum16(float v) {
  for (int off = 1; off < 16; off <<= 1) v += __shfl_xor(v, off, 16);
  return v;
}
#endif

// Stage one 16x32 f32 emission tile (2KB) for step s into LDS.
// 128 x 16B segments, 4 per lane, fully coalesced; async when available.
__device__ __forceinline__ void stage_obs(const float* __restrict__ wtv, int base, int s,
                                          float (*dst)[T_DIM], int lane)
{
#pragma unroll
  for (int i = 0; i < 4; ++i) {
    int flat = i * 32 + lane;          // 0..127
    int row  = flat >> 3;              // 0..15
    int seg  = flat & 7;               // 16B segment within the 128B row
    const float* g = wtv + ((size_t)(base + row) * L_DIM + s) * T_DIM + seg * 4;
#if HAVE_ASYNC
    __builtin_amdgcn_global_load_async_to_lds_b128(
        (gl_i32x4*)(float*)g,                    // strip const, cast pointee+AS1
        (ls_i32x4*)&dst[row][seg * 4],           // generic -> AS3 b128
        0, 0);
#else
    *(float4*)&dst[row][seg * 4] = *(const float4*)g;
#endif
  }
}

// Row-wise max over the 3 N-tiles (DPP reduce), update carried normalizer cc,
// then store P = exp2(d - m) into the wave-private LDS tile (row-major
// [16][64] f16) for the A-fragment transpose.  d/m/cc are in log2 units.
__device__ __forceinline__ void normalize_store(const v8f d[3], float cc[8], float m[8],
                                                _Float16 (*ldsP)[64], int hf, int nn)
{
#pragma unroll
  for (int r = 0; r < 8; ++r) {
    m[r] = redmax16(fmaxf(fmaxf(d[0][r], d[1][r]), d[2][r]));
    cc[r] += m[r];
  }
#pragma unroll
  for (int t = 0; t < 3; ++t) {
#pragma unroll
    for (int r = 0; r < 8; ++r) {
      float p = EXP2F(d[t][r] - m[r]);               // pads: exp2(-1e30 - m) == 0
      ldsP[r + 8 * hf][16 * t + nn] = (_Float16)p;   // C/D layout -> row-major LDS
    }
  }
}

// Read P back from LDS in the exact 16-bit A-matrix 16x32 fragment layout:
// lane holds row (lane&15); K = 16*(e>>3) + 8*(lane>>4) + (e&7), + 32*kt.
// The 8 consecutive (e&7) halves merge into ds_load_b128s.
__device__ __forceinline__ void build_a(v16h a[2], const _Float16 (*ldsP)[64], int hf, int nn)
{
#pragma unroll
  for (int kt = 0; kt < 2; ++kt) {
#pragma unroll
    for (int e = 0; e < 16; ++e) {
      int K = 32 * kt + ((e >> 3) << 4) + (hf << 3) + (e & 7);
      a[kt][e] = ldsP[nn][K];
    }
  }
}

__global__ __launch_bounds__(32) void crf_forward_wmma(const float* __restrict__ wtv,
                                                       const float* __restrict__ trans,
                                                       float* __restrict__ out)
{
  __shared__ _Float16 ldsP[16][64];          // 2 KB transpose tile
  __shared__ float obsb[2][16][T_DIM];       // 4 KB double-buffered emissions
  const int lane = threadIdx.x;              // 0..31
  const int hf   = lane >> 4;                // half: 0 or 1
  const int nn   = lane & 15;
  const int base = blockIdx.x * 16;          // 16 batch rows per wave

  // K-pad columns 48..62 are zero; column 63 is the bias column (always 1.0)
  // feeding the 2^-14 floor row of E.  Written once, never touched again.
#pragma unroll
  for (int i = 0; i < 8; ++i) {
    int col = 48 + hf * 8 + i;
    ldsP[nn][col] = (_Float16)((col == 63) ? 1.0f : 0.0f);
  }

  // Per-lane state-column constants (branchless masking instead of control flow).
  int idxc[3]; int negm[3]; int padm[3];
#pragma unroll
  for (int t = 0; t < 3; ++t) {
    int cur = 16 * t + nn;
    int ix  = cur - 1;
    idxc[t] = ix < 0 ? 0 : (ix > T_DIM - 1 ? T_DIM - 1 : ix);
    negm[t] = (cur == 0) | (cur == S_DIM - 1);
    padm[t] = cur >= S_DIM;
  }

  // Constant E = exp(translation), zero-padded to 64x48, directly in the
  // 16-bit B-matrix 32x16 fragment layout: lane -> N = nn, K = 16*(e>>3) + 8*hf + (e&7).
  // Row 63 = 2^-14 floor (paired with ldsP bias column) => D >= 2^-14 always.
  v16h E[2][3];
#pragma unroll
  for (int kt = 0; kt < 2; ++kt) {
#pragma unroll
    for (int nt = 0; nt < 3; ++nt) {
#pragma unroll
      for (int e = 0; e < 16; ++e) {
        int prev = 32 * kt + ((e >> 3) << 4) + (hf << 3) + (e & 7);
        int cur  = 16 * nt + nn;
        float v = 0.0f;
        if (prev < S_DIM && cur < S_DIM) v = __expf(trans[prev * S_DIM + cur]);
        if (prev == 63 && cur < S_DIM)   v = FLOORH;
        E[kt][nt][e] = (_Float16)v;    // exp(-1000) underflows to exact 0
      }
    }
  }

  // alpha0 in C/D layout (reg r -> batch row r+8*hf, lane -> state 16*t+nn),
  // in log2 units: alpha0 = pad(emissions[:,0,:], 0.0 at START/STOP) * log2e.
  v8f d[3];
  float cc[8], m[8];
#pragma unroll
  for (int r = 0; r < 8; ++r) cc[r] = 0.0f;
#pragma unroll
  for (int t = 0; t < 3; ++t) {
#pragma unroll
    for (int r = 0; r < 8; ++r) {
      float ld = wtv[((size_t)(base + r + 8 * hf) * L_DIM) * T_DIM + idxc[t]];
      float v  = negm[t] ? 0.0f : ld * LOG2E;
      d[t][r]  = padm[t] ? PAD_V : v;
    }
  }

  normalize_store(d, cc, m, ldsP, hf, nn);
  __syncthreads();                       // single-wave WG: cheap, orders LDS
  v16h a[2];
  build_a(a, ldsP, hf, nn);
  __syncthreads();

  // One recursion step: 6 WMMAs, wait for this step's staged tile, then
  // d = obs*log2e + log2(D), renormalize, rebuild the A fragment.  `last`
  // is a literal at each call site so the wait selection constant-folds.
  auto step_core = [&](int s, bool last) __attribute__((always_inline)) {
    const int buf = s & 1;
    v8f acc[3];
#pragma unroll
    for (int nt = 0; nt < 3; ++nt) {
      v8f c = {};
      c = __builtin_amdgcn_wmma_f32_16x16x32_f16(false, a[0], false, E[0][nt],
                                                 (short)0, c, false, false);
      c = __builtin_amdgcn_wmma_f32_16x16x32_f16(false, a[1], false, E[1][nt],
                                                 (short)0, c, false, false);
      acc[nt] = c;
    }
    if (last) { WAIT_ASYNC_0(); } else { WAIT_ASYNC_4(); }
#pragma unroll
    for (int t = 0; t < 3; ++t) {
#pragma unroll
      for (int r = 0; r < 8; ++r) {
        float ld = obsb[buf][r + 8 * hf][idxc[t]];
        float o  = negm[t] ? NEG_V : ld;
        float v  = fmaf(o, LOG2E, LOG2F(acc[t][r]));  // D >= 2^-14: log2 finite
        d[t][r]  = padm[t] ? PAD_V : v;
      }
    }
    normalize_store(d, cc, m, ldsP, hf, nn);
    __syncthreads();
    build_a(a, ldsP, hf, nn);
    __syncthreads();
  };

  // Prologue: kick off async staging of step 1's emission tile.
  stage_obs(wtv, base, 1, obsb[1], lane);

  // Steady state: always stage s+1, always wait asynccnt<=4 (this step's 4
  // transfers were issued before the next step's 4).
#pragma unroll 1
  for (int s = 1; s < L_DIM - 1; ++s) {
    stage_obs(wtv, base, s + 1, obsb[(s + 1) & 1], lane);
    step_core(s, false);
  }
  // Peeled final step: no further staging, drain the async counter.
  step_core(L_DIM - 1, true);

  // logZ[b] = ln2 * (c + log2(sum_cur exp2(alpha_T[cur] - c))); pads give 0.
#pragma unroll
  for (int r = 0; r < 8; ++r) {
    float s8 = EXP2F(d[0][r] - m[r]) + EXP2F(d[1][r] - m[r]) + EXP2F(d[2][r] - m[r]);
    float sum = redsum16(s8);
    if (nn == 0) out[base + r + 8 * hf] = (cc[r] + LOG2F(sum)) * LN2;
  }
}

// Gold-path score: one wave per batch row, lanes parallel over L, shuffle
// reduction. Runs after the forward kernel on the same stream and turns
// out[b] (=logZ) into the final loss logZ - score_path.
__global__ __launch_bounds__(256) void crf_path_sub(const float* __restrict__ wtv,
                                                    const float* __restrict__ trans,
                                                    const int* __restrict__ tags,
                                                    float* __restrict__ out)
{
  const int b    = (blockIdx.x * blockDim.x + threadIdx.x) >> 5;
  const int lane = threadIdx.x & 31;
  if (b >= B_DIM) return;
  const int* tg = tags + (size_t)b * L_DIM;

  float sc = 0.0f;
#pragma unroll 1
  for (int j = lane; j < L_DIM; j += 32) {
    int t = tg[j];
    sc += wtv[((size_t)b * L_DIM + j) * T_DIM + (t - 1)];      // gold emission
    if (j + 1 < L_DIM) sc += trans[t * S_DIM + tg[j + 1]];     // gold transition
  }
#pragma unroll
  for (int off = 16; off > 0; off >>= 1) sc += __shfl_xor(sc, off, 32);
  if (lane == 0) {
    sc += trans[tg[0]]                                         // translation[0, tag0]
        + trans[tg[L_DIM - 1] * S_DIM + (S_DIM - 1)];          // translation[last, STOP]
    out[b] -= sc;
  }
}

extern "C" void kernel_launch(void* const* d_in, const int* in_sizes, int n_in,
                              void* d_out, int out_size, void* d_ws, size_t ws_size,
                              hipStream_t stream)
{
  const float* wtv   = (const float*)d_in[0];   // [512,512,32] f32
  const float* trans = (const float*)d_in[1];   // [34,34] f32
  const int*   tags  = (const int*)d_in[2];     // [512,512] i32
  float* out = (float*)d_out;                   // [512] f32

  crf_forward_wmma<<<dim3(B_DIM / 16), dim3(32), 0, stream>>>(wtv, trans, out);
  crf_path_sub<<<dim3((B_DIM * 32) / 256), dim3(256), 0, stream>>>(wtv, trans, tags, out);
}
